// DMM_42691974922534
// MI455X (gfx1250) — matrix operand order repaired and checked by reading, not compiled, not run on hardware
//
#include <hip/hip_runtime.h>
#include <hip/hip_bf16.h>

typedef __attribute__((ext_vector_type(16))) _Float16 v16h;
typedef __attribute__((ext_vector_type(8)))  _Float16 v8h;
typedef __attribute__((ext_vector_type(8)))  float    v8f;

static constexpr int BB = 128, TT = 512, INW = 88, ZW = 80, HW = 400, EW = 100, TRW = 200;
static constexpr int BT = BB * TT;                 // 65536 rows
static constexpr int KP_IN = 96, KP_H = 416, KP_Z = 96, KP_TR = 224, KP_E = 128;
static constexpr int OUTW = INW + ZW + ZW;         // 248

// ---------------- math helpers ----------------
__device__ __forceinline__ float sigmf(float x) { return 1.f / (1.f + __expf(-x)); }
__device__ __forceinline__ float splusf(float x) { return (x > 20.f) ? x : log1pf(__expf(x)); }

// ---------------- WMMA fragment helpers (wave32, v_wmma_f32_16x16x32_f16) ----
__device__ __forceinline__ v16h cat8(v8h a, v8h b) {
  return __builtin_shufflevector(a, b, 0,1,2,3,4,5,6,7,8,9,10,11,12,13,14,15);
}
// A (16xK, row per lane%16): lane<16 -> K = k0+0..7 | k0+16..23 ; lane>=16 -> +8
__device__ __forceinline__ v16h frag_a(const _Float16* base, int pitch, int k0, int lane) {
  int row = lane & 15;
  int off = k0 + ((lane & 16) ? 8 : 0);
  const _Float16* p = base + (size_t)row * pitch + off;
  v8h lo = *(const v8h*)(p);
  v8h hi = *(const v8h*)(p + 16);
  return cat8(lo, hi);
}
// B stored as Bt[n][Kpad] (i.e. W row-major). lane<16 -> N=n0+l, K=k0+0..7 | k0+16..23 ; lane>=16 -> +8
__device__ __forceinline__ v16h frag_b(const _Float16* bt, int pitch, int n0, int k0, int lane, int nmax) {
  int n = n0 + (lane & 15);
  n = n > nmax ? nmax : n;
  int off = k0 + ((lane & 16) ? 8 : 0);
  const _Float16* p = bt + (size_t)n * pitch + off;
  v8h lo = *(const v8h*)(p);
  v8h hi = *(const v8h*)(p + 16);
  return cat8(lo, hi);
}
__device__ __forceinline__ v8f wmma16(v16h a, v16h b, v8f c) {
  return __builtin_amdgcn_wmma_f32_16x16x32_f16(false, a, false, b, (short)0, c, false, false);
}

// ---------------- prep kernels ----------------
// f32 weight [N][K] -> f16 [N][Kp], zero-padded K
__global__ void convw(const float* __restrict__ src, _Float16* __restrict__ dst, int N, int K, int Kp) {
  int i = blockIdx.x * 256 + threadIdx.x;
  if (i >= N * Kp) return;
  int n = i / Kp, k = i - n * Kp;
  dst[i] = (k < K) ? (_Float16)src[n * K + k] : (_Float16)0.f;
}
// mini_batch [B][T][88] f32 -> xrA [b*T+s][96] f16 with time reversed (s -> t=511-s)
__global__ void convx(const float* __restrict__ x, _Float16* __restrict__ dst) {
  int i = blockIdx.x * 256 + threadIdx.x;
  if (i >= BT * KP_IN) return;
  int row = i / KP_IN, k = i - row * KP_IN;
  int b = row >> 9, s = row & 511;
  _Float16 v = (_Float16)0.f;
  if (k < INW) v = (_Float16)x[((size_t)b * TT + (TT - 1 - s)) * INW + k];
  dst[i] = v;
}
// zero the pad columns [nvalid, pitch) of an f16 activation buffer (BT rows)
__global__ void padzero(_Float16* __restrict__ buf, int pitch, int nvalid, int total) {
  int i = blockIdx.x * 256 + threadIdx.x;
  if (i >= total) return;
  int padw = pitch - nvalid;
  int r = i / padw, c = nvalid + (i - r * padw);
  buf[(size_t)r * pitch + c] = (_Float16)0.f;
}
// zshiftA[b*T+t] = (t==0) ? z0 : zA[b*T+t-1]   (96 cols, pads flow through as zero)
__global__ void shiftz(const _Float16* __restrict__ zA, const float* __restrict__ z0,
                       _Float16* __restrict__ dst) {
  int i = blockIdx.x * 256 + threadIdx.x;
  if (i >= BT * KP_Z) return;
  int row = i / KP_Z, k = i - row * KP_Z;
  int t = row & 511;
  _Float16 v;
  if (t == 0) v = (k < ZW) ? (_Float16)z0[k] : (_Float16)0.f;
  else        v = zA[(size_t)(row - 1) * KP_Z + k];
  dst[i] = v;
}

// ---------------- generic WMMA GEMM: C[BT x N] = act(A[BT x K] * Bt[N x K]^T + bias) ----
// EPI 0: f32 store | 1: relu->f16 | 2: sigmoid->f32 | 3: softplus->f32
// EPI 4: z = s0 + softplus(val)*s1 -> f16 | 5: f32 store + relu->f16 (d1h) | 6: (1-s0)*val + s0*s1 -> f32
template<int EPI>
__global__ void __launch_bounds__(128) gemm16(
    const _Float16* __restrict__ A, const _Float16* __restrict__ Bt,
    const float* __restrict__ bias, int N, int K,
    float* __restrict__ d0f, _Float16* __restrict__ d0h, int d0p, int d0c,
    _Float16* __restrict__ d1h, int d1p,
    const float* __restrict__ s0, int s0p, const float* __restrict__ s1, int s1p) {
  __shared__ _Float16 As[16 * KP_H];  // max K = 416
  int m0 = blockIdx.x << 4;
  int cnt4 = (16 * K) >> 3;
  const uint4* Ag = (const uint4*)(A + (size_t)m0 * K);
  uint4* As4 = (uint4*)As;
  for (int i = threadIdx.x; i < cnt4; i += 128) As4[i] = Ag[i];
  __syncthreads();
  int lane = threadIdx.x & 31, wave = threadIdx.x >> 5;
  int ntiles = (N + 15) >> 4;
  int nkt = K >> 5;
  for (int nt = wave; nt < ntiles; nt += 4) {
    int n0 = nt << 4;
    v8f acc = {};
    for (int kt = 0; kt < nkt; ++kt) {
      v16h a = frag_a(As, K, kt << 5, lane);
      v16h b = frag_b(Bt, K, n0, kt << 5, lane, N - 1);
      acc = wmma16(a, b, acc);
    }
    int col = n0 + (lane & 15);
    if (col >= N) continue;
    float bb = bias[col];
    int mb = (lane & 16) ? 8 : 0;
#pragma unroll
    for (int r = 0; r < 8; ++r) {
      size_t row = (size_t)(m0 + mb + r);
      float val = acc[r] + bb;
      if constexpr (EPI == 0) {
        d0f[row * d0p + d0c + col] = val;
      } else if constexpr (EPI == 1) {
        d0h[row * d0p + d0c + col] = (_Float16)(val > 0.f ? val : 0.f);
      } else if constexpr (EPI == 2) {
        d0f[row * d0p + d0c + col] = sigmf(val);
      } else if constexpr (EPI == 3) {
        d0f[row * d0p + d0c + col] = splusf(val);
      } else if constexpr (EPI == 4) {
        float z = s0[row * s0p + col] + splusf(val) * s1[row * s1p + col];
        d0h[row * d0p + d0c + col] = (_Float16)z;
      } else if constexpr (EPI == 5) {
        d0f[row * d0p + d0c + col] = val;
        d1h[row * d1p + col] = (_Float16)(val > 0.f ? val : 0.f);
      } else if constexpr (EPI == 6) {
        float g = s0[row * s0p + col];
        float p = s1[row * s1p + col];
        d0f[row * d0p + d0c + col] = (1.f - g) * val + g * p;
      }
    }
  }
}

// ---------------- recurrent scan ----------------
// 8 workgroups x 16 batch rows, 32 waves each (1024 threads, wave32). Each wave owns exactly ONE
// of the 25 N-tiles, so the per-step critical path is a single 13-deep WMMA accumulator chain.
// W_hh K-tiles 0..11 live in LDS (307200 B); K-tail (k=384..415, zero-padded) streams from L2.
// h ping-pongs through a 16x416 f16 LDS buffer. h_s = relu(pre[:,s,:] + h_{s-1} @ W_hh^T + b_hh);
// rnn_out written at t = 511-s. pre rows for step s+1 are prefetched during step s.
__global__ void __launch_bounds__(1024) rnn_scan(
    const float* __restrict__ pre,          // [BT][400], s-order
    const _Float16* __restrict__ whh,       // [400][416] f16, zero-padded K
    const float* __restrict__ b_hh, const float* __restrict__ h0,
    _Float16* __restrict__ rnnA) {          // [BT][416] f16, t-order, pads pre-zeroed
  extern __shared__ _Float16 smem[];
  _Float16* wl = smem;                 // [400][384]
  _Float16* hb = smem + 400 * 384;     // [16][416]
  int tid = threadIdx.x, lane = tid & 31, wave = tid >> 5;
  int b0 = blockIdx.x << 4;
  // stage W_hh K<384 into LDS (rows of 48 uint4 from rows of 52 uint4)
  for (int i = tid; i < 400 * 48; i += 1024) {
    int r = i / 48, c = i - r * 48;
    ((uint4*)wl)[i] = ((const uint4*)whh)[r * 52 + c];
  }
  // init h buffer
  for (int i = tid; i < 16 * KP_H; i += 1024) {
    int k = i % KP_H;
    hb[i] = (k < HW) ? (_Float16)h0[k] : (_Float16)0.f;
  }
  __syncthreads();
  const bool active = (wave < 25);          // wave-uniform: EXEC all-ones inside WMMA region
  int n0 = wave << 4;
  int col = n0 + (lane & 15);
  int mb = (lane & 16) ? 8 : 0;
  float bh = active ? b_hh[col] : 0.f;
  for (int s = 0; s < TT; ++s) {
    float hres[8] = {};
    if (active) {
      v8f acc = {};
      for (int kt = 0; kt < 13; ++kt) {
        v16h a = frag_a(hb, KP_H, kt << 5, lane);
        v16h b = (kt < 12) ? frag_b(wl, 384, n0, kt << 5, lane, 399)
                           : frag_b(whh, KP_H, n0, kt << 5, lane, 399);
        acc = wmma16(a, b, acc);
      }
#pragma unroll
      for (int r = 0; r < 8; ++r) {
        size_t brow = (size_t)(b0 + mb + r);
        float v = acc[r] + pre[(brow * TT + s) * HW + col] + bh;
        hres[r] = v > 0.f ? v : 0.f;
        if (s + 1 < TT)  // hide next step's only HBM dependency (global_prefetch_b8)
          __builtin_prefetch(&pre[(brow * TT + s + 1) * HW + col], 0, 1);
      }
    }
    __syncthreads();  // all waves done reading h_{s-1}
    if (active) {
#pragma unroll
      for (int r = 0; r < 8; ++r) {
        _Float16 hv = (_Float16)hres[r];
        hb[(mb + r) * KP_H + col] = hv;
        rnnA[((size_t)(b0 + mb + r) * TT + (TT - 1 - s)) * KP_H + col] = hv;
      }
    }
    __syncthreads();  // h_s visible before next step
  }
}

// ---------------- launcher ----------------
extern "C" void kernel_launch(void* const* d_in, const int* in_sizes, int n_in,
                              void* d_out, int out_size, void* d_ws, size_t ws_size,
                              hipStream_t stream) {
  (void)in_sizes; (void)n_in; (void)out_size; (void)ws_size;
  const float* mini_batch = (const float*)d_in[0];
  const float* eps   = (const float*)d_in[1];
  const float* W_ih  = (const float*)d_in[2];
  const float* W_hh  = (const float*)d_in[3];
  const float* b_ih  = (const float*)d_in[4];
  const float* b_hh  = (const float*)d_in[5];
  const float* h0    = (const float*)d_in[6];
  const float* z0    = (const float*)d_in[7];
  const float* W_loc = (const float*)d_in[8];
  const float* b_loc = (const float*)d_in[9];
  const float* W_scl = (const float*)d_in[10];
  const float* b_scl = (const float*)d_in[11];
  const float* gW1 = (const float*)d_in[12]; const float* gb1 = (const float*)d_in[13];
  const float* gW2 = (const float*)d_in[14]; const float* gb2 = (const float*)d_in[15];
  const float* pW1 = (const float*)d_in[16]; const float* pb1 = (const float*)d_in[17];
  const float* pW2 = (const float*)d_in[18]; const float* pb2 = (const float*)d_in[19];
  const float* sW = (const float*)d_in[20];  const float* sb = (const float*)d_in[21];
  const float* lW = (const float*)d_in[22];  const float* lb = (const float*)d_in[23];
  const float* eW1 = (const float*)d_in[24]; const float* eb1 = (const float*)d_in[25];
  const float* eW2 = (const float*)d_in[26]; const float* eb2 = (const float*)d_in[27];
  const float* eW3 = (const float*)d_in[28]; const float* eb3 = (const float*)d_in[29];
  float* out = (float*)d_out;

  char* ws = (char*)d_ws;
  size_t off = 0;
  auto alloc = [&](size_t bytes) { char* p = ws + off; off = (off + bytes + 255) & ~(size_t)255; return p; };
  _Float16* xrA   = (_Float16*)alloc((size_t)BT * KP_IN * 2);
  _Float16* wIH   = (_Float16*)alloc((size_t)HW * KP_IN * 2);
  _Float16* wHH   = (_Float16*)alloc((size_t)HW * KP_H * 2);
  _Float16* wLOC  = (_Float16*)alloc((size_t)ZW * KP_H * 2);
  _Float16* wSCL  = (_Float16*)alloc((size_t)ZW * KP_H * 2);
  _Float16* wG1   = (_Float16*)alloc((size_t)TRW * KP_Z * 2);
  _Float16* wG2   = (_Float16*)alloc((size_t)ZW * KP_TR * 2);
  _Float16* wP1   = (_Float16*)alloc((size_t)TRW * KP_Z * 2);
  _Float16* wP2   = (_Float16*)alloc((size_t)ZW * KP_TR * 2);
  _Float16* wS    = (_Float16*)alloc((size_t)ZW * KP_Z * 2);
  _Float16* wL    = (_Float16*)alloc((size_t)ZW * KP_Z * 2);
  _Float16* wE1   = (_Float16*)alloc((size_t)EW * KP_Z * 2);
  _Float16* wE2   = (_Float16*)alloc((size_t)EW * KP_E * 2);
  _Float16* wE3   = (_Float16*)alloc((size_t)INW * KP_E * 2);
  float*    pre   = (float*)   alloc((size_t)BT * HW * 4);
  _Float16* rnnA  = (_Float16*)alloc((size_t)BT * KP_H * 2);
  float*    zloc  = (float*)   alloc((size_t)BT * ZW * 4);
  _Float16* zA    = (_Float16*)alloc((size_t)BT * KP_Z * 2);
  _Float16* zsA   = (_Float16*)alloc((size_t)BT * KP_Z * 2);
  _Float16* hidG  = (_Float16*)alloc((size_t)BT * KP_TR * 2);
  _Float16* hidP  = (_Float16*)alloc((size_t)BT * KP_TR * 2);
  float*    gate  = (float*)   alloc((size_t)BT * ZW * 4);
  float*    prop  = (float*)   alloc((size_t)BT * ZW * 4);
  _Float16* prA   = (_Float16*)alloc((size_t)BT * KP_Z * 2);
  _Float16* e1A   = (_Float16*)alloc((size_t)BT * KP_E * 2);
  _Float16* e2A   = (_Float16*)alloc((size_t)BT * KP_E * 2);

  auto blk = [](int n) { return dim3((n + 255) / 256); };
  // zero pad columns of every f16 activation buffer used as a WMMA A operand
  padzero<<<blk(BT * (KP_H  - HW)), 256, 0, stream>>>(rnnA, KP_H,  HW,  BT * (KP_H  - HW));
  padzero<<<blk(BT * (KP_Z  - ZW)), 256, 0, stream>>>(zA,   KP_Z,  ZW,  BT * (KP_Z  - ZW));
  padzero<<<blk(BT * (KP_TR - TRW)),256, 0, stream>>>(hidG, KP_TR, TRW, BT * (KP_TR - TRW));
  padzero<<<blk(BT * (KP_TR - TRW)),256, 0, stream>>>(hidP, KP_TR, TRW, BT * (KP_TR - TRW));
  padzero<<<blk(BT * (KP_Z  - ZW)), 256, 0, stream>>>(prA,  KP_Z,  ZW,  BT * (KP_Z  - ZW));
  padzero<<<blk(BT * (KP_E  - EW)), 256, 0, stream>>>(e1A,  KP_E,  EW,  BT * (KP_E  - EW));
  padzero<<<blk(BT * (KP_E  - EW)), 256, 0, stream>>>(e2A,  KP_E,  EW,  BT * (KP_E  - EW));
  // weight conversion f32 -> padded f16
  convw<<<blk(HW * KP_IN), 256, 0, stream>>>(W_ih, wIH, HW, INW, KP_IN);
  convw<<<blk(HW * KP_H),  256, 0, stream>>>(W_hh, wHH, HW, HW,  KP_H);
  convw<<<blk(ZW * KP_H),  256, 0, stream>>>(W_loc, wLOC, ZW, HW, KP_H);
  convw<<<blk(ZW * KP_H),  256, 0, stream>>>(W_scl, wSCL, ZW, HW, KP_H);
  convw<<<blk(TRW * KP_Z), 256, 0, stream>>>(gW1, wG1, TRW, ZW, KP_Z);
  convw<<<blk(ZW * KP_TR), 256, 0, stream>>>(gW2, wG2, ZW, TRW, KP_TR);
  convw<<<blk(TRW * KP_Z), 256, 0, stream>>>(pW1, wP1, TRW, ZW, KP_Z);
  convw<<<blk(ZW * KP_TR), 256, 0, stream>>>(pW2, wP2, ZW, TRW, KP_TR);
  convw<<<blk(ZW * KP_Z),  256, 0, stream>>>(sW, wS, ZW, ZW, KP_Z);
  convw<<<blk(ZW * KP_Z),  256, 0, stream>>>(lW, wL, ZW, ZW, KP_Z);
  convw<<<blk(EW * KP_Z),  256, 0, stream>>>(eW1, wE1, EW, ZW, KP_Z);
  convw<<<blk(EW * KP_E),  256, 0, stream>>>(eW2, wE2, EW, EW, KP_E);
  convw<<<blk(INW * KP_E), 256, 0, stream>>>(eW3, wE3, INW, EW, KP_E);
  convx<<<blk(BT * KP_IN), 256, 0, stream>>>(mini_batch, xrA);

#define GEMM(EPI, A_, B_, BIAS_, N_, K_, D0F_, D0H_, D0P_, D0C_, D1H_, D1P_, S0_, S0P_, S1_, S1P_) \
  gemm16<EPI><<<dim3(BT / 16), 128, 0, stream>>>(A_, B_, BIAS_, N_, K_, D0F_, D0H_, D0P_, D0C_, \
                                                 D1H_, D1P_, S0_, S0P_, S1_, S1P_)
  // pre = xr @ W_ih^T + b_ih   (s-order)
  GEMM(0, xrA, wIH, b_ih, HW, KP_IN, pre, nullptr, HW, 0, nullptr, 0, nullptr, 0, nullptr, 0);
  // recurrent scan (writes rnnA in t-order)
  rnn_scan<<<dim3(BB / 16), 1024, (400 * 384 + 16 * KP_H) * sizeof(_Float16), stream>>>(
      pre, wHH, b_hh, h0, rnnA);
  // z_loc, z = z_loc + softplus(.)*eps
  GEMM(0, rnnA, wLOC, b_loc, ZW, KP_H, zloc, nullptr, ZW, 0, nullptr, 0, nullptr, 0, nullptr, 0);
  GEMM(4, rnnA, wSCL, b_scl, ZW, KP_H, nullptr, zA, KP_Z, 0, nullptr, 0, zloc, ZW, eps, ZW);
  shiftz<<<blk(BT * KP_Z), 256, 0, stream>>>(zA, z0, zsA);
  // gate / prop MLPs
  GEMM(1, zsA, wG1, gb1, TRW, KP_Z, nullptr, hidG, KP_TR, 0, nullptr, 0, nullptr, 0, nullptr, 0);
  GEMM(2, hidG, wG2, gb2, ZW, KP_TR, gate, nullptr, ZW, 0, nullptr, 0, nullptr, 0, nullptr, 0);
  GEMM(1, zsA, wP1, pb1, TRW, KP_Z, nullptr, hidP, KP_TR, 0, nullptr, 0, nullptr, 0, nullptr, 0);
  GEMM(5, hidP, wP2, pb2, ZW, KP_TR, prop, nullptr, ZW, 0, prA, KP_Z, nullptr, 0, nullptr, 0);
  // trans_scale = softplus(relu(prop) @ sW^T + sb) -> out[:,168:248]
  GEMM(3, prA, wS, sb, ZW, KP_Z, out, nullptr, OUTW, INW + ZW, nullptr, 0, nullptr, 0, nullptr, 0);
  // trans_loc = (1-gate)*(z_shift @ lW^T + lb) + gate*prop -> out[:,88:168]
  GEMM(6, zsA, wL, lb, ZW, KP_Z, out, nullptr, OUTW, INW, nullptr, 0, gate, ZW, prop, ZW);
  // emission MLP -> out[:,0:88]
  GEMM(1, zA, wE1, eb1, EW, KP_Z, nullptr, e1A, KP_E, 0, nullptr, 0, nullptr, 0, nullptr, 0);
  GEMM(1, e1A, wE2, eb2, EW, KP_E, nullptr, e2A, KP_E, 0, nullptr, 0, nullptr, 0, nullptr, 0);
  GEMM(2, e2A, wE3, eb3, INW, KP_E, out, nullptr, OUTW, 0, nullptr, 0, nullptr, 0, nullptr, 0);
#undef GEMM
}
